// SCSSD_15504831938725
// MI455X (gfx1250) — matrix operand order repaired and checked by reading, not compiled, not run on hardware
//
#include <hip/hip_runtime.h>
#include <cstdint>

// ---------------- problem constants ----------------
#define B0     8
#define CIN    256
#define HW     9216            // 96*96 = L
#define SPLIT  8
#define DM     32              // CIN/SPLIT
#define DIN    64              // EXPAND*DM
#define DSTATE 64
#define NH     8
#define HD     8               // head dim
#define DOUT   256
#define CHUNK  32
#define NCHUNK 288             // HW/CHUNK
#define DPROJ  264             // 2*DIN + 2*DSTATE + NH
#define DPROJ_PAD 272          // padded to 17 tiles of 16
#define CONV_CH 192            // DIN + 2*DSTATE
#define NB     64              // SPLIT*B0

typedef unsigned int u32;
typedef __attribute__((ext_vector_type(16))) __bf16 v16bf;
typedef __attribute__((ext_vector_type(8)))  float  v8f;
typedef __attribute__((ext_vector_type(4)))  u32    v4u;
typedef __attribute__((ext_vector_type(8)))  int    v8i;
typedef __attribute__((ext_vector_type(4)))  int    v4i;

#if defined(__gfx1250__) && __has_builtin(__builtin_amdgcn_tensor_load_to_lds) && \
    __has_builtin(__builtin_amdgcn_s_wait_tensorcnt)
#define USE_TDM 1
#else
#define USE_TDM 0
#endif

// ---------------- WMMA helpers (CDNA5 16x16x32 bf16, ISA 7.12.2 layouts) ----
__device__ __forceinline__ v8f wmma_bf16(v16bf a, v16bf b, v8f c) {
  return __builtin_amdgcn_wmma_f32_16x16x32_bf16(false, a, false, b, (short)0, c,
                                                 false, false);
}
// A fragment (16xK, 16-bit): lane holds row M=lane&15
__device__ __forceinline__ int a_row(int lane)        { return lane & 15; }
__device__ __forceinline__ int a_col(int lane, int i) { return ((lane >> 4) << 3) + (i < 8 ? i : i + 8); }
// B fragment (Kx16, 16-bit): lane holds col N=lane&15
__device__ __forceinline__ int b_row(int lane, int i) { return ((lane >> 4) << 3) + (i < 8 ? i : i + 8); }
__device__ __forceinline__ int b_col(int lane)        { return lane & 15; }
// C/D f32 fragment
__device__ __forceinline__ int d_row(int lane, int r) { return ((lane >> 4) << 3) + r; }
__device__ __forceinline__ int d_col(int lane)        { return lane & 15; }
// producer-side: element (m,k) of a 16x32 A-tile -> flat position lane*16+i
__device__ __forceinline__ int afrag_pos(int m, int k) {
  int lane = m + (((k >> 3) & 1) << 4);
  int i = (k & 7) + ((k >> 4) << 3);
  return lane * 16 + i;
}

__device__ __forceinline__ float warp_allreduce_sum(float v) {
  #pragma unroll
  for (int off = 16; off > 0; off >>= 1) v += __shfl_xor(v, off, 32);
  return v;
}
__device__ __forceinline__ float silu_f(float x) { return x / (1.f + __expf(-x)); }

#if USE_TDM
// Contiguous 1D f32 global -> LDS DMA via Tensor Data Mover (D# per ISA 8.3/8.4).
// This toolchain's builtin takes 6 args: (g0, g1, g2, g3, g2b, cpol).
__device__ __forceinline__ void tdm_load_1d_f32(u32 lds_addr, const float* g, int nelem) {
  unsigned long long ga = (unsigned long long)(uintptr_t)g;
  v4u g0;
  g0.x = 1u;                                            // count=1, user descriptor
  g0.y = lds_addr;                                      // LDS byte address
  g0.z = (u32)ga;                                       // global_addr[31:0]
  g0.w = (u32)((ga >> 32) & 0x01FFFFFFull) | 0x80000000u; // addr[56:32] | type=2
  v8i g1;
  g1[0] = 0x00020000;                                   // data_size=4B, mask=0
  g1[1] = (int)((nelem & 0xFFFF) << 16);                // tensor_dim0[15:0]
  g1[2] = (int)(((u32)nelem >> 16) & 0xFFFF) | (1 << 16); // dim0 hi | tensor_dim1=1
  g1[3] = (int)((nelem & 0xFFFF) << 16);                // tile_dim0 = nelem
  g1[4] = 1;                                            // tile_dim1=1, tile_dim2=0
  g1[5] = nelem;                                        // tensor_dim0_stride lo32
  g1[6] = 0;
  g1[7] = 0;
  v4i z4 = {0, 0, 0, 0};
  v8i z8 = {0, 0, 0, 0, 0, 0, 0, 0};
  __builtin_amdgcn_tensor_load_to_lds(g0, g1, z4, z4, z8, 0);
}
#endif

// ---------------- workspace layout (unchanged sizes, 256B aligned) ----------
static constexpr size_t NBL     = (size_t)NB * HW;       // 589824 tokens
static constexpr size_t OFF_XSF = 0;                                   // f32 [NBL][32]
static constexpr size_t OFF_XSB = OFF_XSF + NBL * DM * 4;              // bf16 A-frags
static constexpr size_t OFF_ZX  = OFF_XSB + NBL * DM * 2;              // f32 [NBL][264]
static constexpr size_t OFF_XH  = OFF_ZX  + NBL * DPROJ * 4;           // f32 [NBL][64]
static constexpr size_t OFF_BC  = OFF_XH  + NBL * DIN * 4;             // f32 [NBL][64]
static constexpr size_t OFF_CC  = OFF_BC  + NBL * DSTATE * 4;          // f32 [NBL][64]
static constexpr size_t OFF_DT  = OFF_CC  + NBL * DSTATE * 4;          // f32 [NBL][8]
static constexpr size_t OFF_DA  = OFF_DT  + NBL * NH * 4;              // f32 [NBL][8]
static constexpr size_t OFF_YS  = OFF_DA  + NBL * NH * 4;              // f32 [NBL][64]
static constexpr size_t OFF_YBF = OFF_YS  + NBL * DIN * 4;             // bf16 A-frags
static constexpr size_t OFF_YMC = OFF_YBF + NBL * DIN * 2;             // f32 [B0*HW][256]
static constexpr size_t OFF_YNB = OFF_YMC + (size_t)B0 * HW * CIN * 4; // bf16 A-frags
static constexpr size_t OFF_WIN = OFF_YNB + (size_t)B0 * HW * CIN * 2; // bf16 B-frags
static constexpr size_t OFF_WOU = OFF_WIN + 32 * DPROJ_PAD * 2;        // bf16 B-frags
static constexpr size_t OFF_PW  = OFF_WOU + DIN * DM * 2;              // bf16 B-frags

// ---------------- K0: weights -> bf16 B-fragment order ----------------------
__global__ void k0_prep(const float* __restrict__ W_in, const float* __restrict__ W_out,
                        const float* __restrict__ proj_w,
                        __bf16* __restrict__ Winb, __bf16* __restrict__ Woutb,
                        __bf16* __restrict__ Pwb) {
  int t = blockIdx.x * blockDim.x + threadIdx.x;          // 0..65535
  if (t < 17 * 512) {              // W_in: 17 N-tiles, K=32 (padded cols -> 0)
    int f = t >> 9, pos = t & 511, lane = pos >> 4, i = pos & 15;
    int k = b_row(lane, i);
    int n = f * 16 + b_col(lane);
    Winb[t] = (__bf16)((n < DPROJ) ? W_in[k * DPROJ + n] : 0.f);
  }
  if (t < 4 * 512) {               // W_out: frag f = ks*2+nt
    int f = t >> 9, pos = t & 511, lane = pos >> 4, i = pos & 15;
    int ks = f >> 1, nt = f & 1;
    int k = b_row(lane, i) + ks * 32;
    int n = nt * 16 + b_col(lane);
    Woutb[t] = (__bf16)W_out[k * DM + n];
  }
  if (t < 128 * 512) {             // proj_w: frag f = nt*8+ks
    int f = t >> 9, pos = t & 511, lane = pos >> 4, i = pos & 15;
    int nt = f >> 3, ks = f & 7;
    int k = b_row(lane, i) + ks * 32;
    int n = nt * 16 + b_col(lane);
    Pwb[t] = (__bf16)proj_w[k * DOUT + n];
  }
}

// ---------------- K1: LayerNorm over C=256 + split; bf16 in A-frag order ----
__global__ void k1_ln1(const float* __restrict__ x, const float* __restrict__ ln_w,
                       const float* __restrict__ ln_b,
                       float* __restrict__ xs_f, __bf16* __restrict__ xs_bf) {
  const int lane = threadIdx.x & 31, wave = threadIdx.x >> 5;
  const int l = blockIdx.x * 8 + wave;   // 0..9215
  const int bb = blockIdx.y;             // 0..7
  float v[8]; float sum = 0.f;
  #pragma unroll
  for (int j = 0; j < 8; ++j) {
    int cc = lane * 8 + j;
    v[j] = x[((long)bb * CIN + cc) * (long)HW + l];
    sum += v[j];
  }
  sum = warp_allreduce_sum(sum);
  const float mu = sum * (1.f / 256.f);
  float sq = 0.f;
  #pragma unroll
  for (int j = 0; j < 8; ++j) { float d = v[j] - mu; sq += d * d; }
  sq = warp_allreduce_sum(sq);
  const float inv = rsqrtf(sq * (1.f / 256.f) + 1e-5f);
  const int s = (lane * 8) >> 5;                          // sequence id (const per lane)
  const long g = ((long)(s * B0 + bb)) * HW + l;          // flat token
  const long T = g >> 4; const int m = (int)(g & 15);
  #pragma unroll
  for (int j = 0; j < 8; ++j) {
    int cc = lane * 8 + j;
    float xn = (v[j] - mu) * inv * ln_w[cc] + ln_b[cc];
    int d = cc & 31;
    xs_f[g * DM + d] = xn;
    xs_bf[T * 512 + afrag_pos(m, d)] = (__bf16)xn;
  }
}

// ---------------- K2: in-proj GEMM [NBL x 32] @ [32 x 272] via WMMA ---------
__global__ void k2_inproj(const __bf16* __restrict__ xs_bf, const __bf16* __restrict__ Winb,
                          float* __restrict__ zx) {
  const int lane = threadIdx.x & 31, wave = threadIdx.x >> 5;
  const long t0 = (long)blockIdx.x * 128 + wave * 16;  // token tile
  const long T0 = t0 >> 4;
  const v16bf a = *(const v16bf*)(xs_bf + T0 * 512 + lane * 16);
  for (int nt = 0; nt < 17; ++nt) {
    const v16bf b = *(const v16bf*)(Winb + nt * 512 + lane * 16);
    v8f acc = {};
    acc = wmma_bf16(a, b, acc);
    const int col = nt * 16 + d_col(lane);
    if (col < DPROJ) {
      #pragma unroll
      for (int r = 0; r < 8; ++r)
        zx[(t0 + d_row(lane, r)) * DPROJ + col] = acc[r];
    }
  }
}

// ---------------- K3: causal depthwise conv(3)+SiLU (rolling regs), dt ------
__global__ void k3_conv(const float* __restrict__ zx, const float* __restrict__ conv_w,
                        const float* __restrict__ conv_b, const float* __restrict__ dt_bias,
                        const float* __restrict__ A_log,
                        float* __restrict__ xh, float* __restrict__ Bc,
                        float* __restrict__ Cc, float* __restrict__ dtb,
                        float* __restrict__ dA) {
  const int c  = threadIdx.x;     // channel (0..191) or head lane (192..199)
  const int bi = blockIdx.y;      // 0..63
  const int l0 = blockIdx.x * 64;
  const long tb = (long)bi * HW;
  if (c < CONV_CH) {
    const float w0 = conv_w[c * 3 + 0], w1 = conv_w[c * 3 + 1], w2 = conv_w[c * 3 + 2];
    const float bcv = conv_b[c];
    float p0 = (l0 >= 2) ? zx[(tb + l0 - 2) * DPROJ + DIN + c] : 0.f;
    float p1 = (l0 >= 1) ? zx[(tb + l0 - 1) * DPROJ + DIN + c] : 0.f;
    for (int l = l0; l < l0 + 64; ++l) {
      const float p2 = zx[(tb + l) * DPROJ + DIN + c];
      const float acc = bcv + w0 * p0 + w1 * p1 + w2 * p2;
      const float o = silu_f(acc);
      if (c < DIN)                 xh[(tb + l) * DIN + c] = o;
      else if (c < DIN + DSTATE)   Bc[(tb + l) * DSTATE + (c - DIN)] = o;
      else                         Cc[(tb + l) * DSTATE + (c - DIN - DSTATE)] = o;
      p0 = p1; p1 = p2;
    }
  } else if (c < CONV_CH + NH) {
    const int h = c - CONV_CH;
    const float Ah = -__expf(A_log[h]);
    const float db = dt_bias[h];
    for (int l = l0; l < l0 + 64; ++l) {
      float xr = zx[(tb + l) * DPROJ + (DPROJ - NH) + h] + db;
      float dt = (xr > 20.f) ? xr : log1pf(__expf(xr));
      dtb[(tb + l) * NH + h] = dt;
      dA[(tb + l) * NH + h]  = dt * Ah;
    }
  }
}

// ---------------- K4: chunked SSD scan; TDM chunk staging; WMMA G=C.B^T -----
__global__ void k4_ssd(const float* __restrict__ Bc, const float* __restrict__ Cc,
                       const float* __restrict__ xh, const float* __restrict__ dtb,
                       const float* __restrict__ dA, float* __restrict__ ys) {
  const int bi = blockIdx.x;       // 0..63
  const int h  = blockIdx.y;       // 0..7
  const int t  = threadIdx.x;      // 0..255
  const int lane = t & 31, wave = t >> 5;

  __shared__ __attribute__((aligned(16))) float sB[CHUNK * DSTATE];
  __shared__ __attribute__((aligned(16))) float sC[CHUNK * DSTATE];
  __shared__ float sx[CHUNK * HD];
  __shared__ float sG[CHUNK * CHUNK];
  __shared__ float sac[CHUNK];
  __shared__ float sS[HD * DSTATE];

  for (int i = t; i < HD * DSTATE; i += 256) sS[i] = 0.f;
  __syncthreads();

  for (int c = 0; c < NCHUNK; ++c) {
    const long base = (long)bi * HW + c * CHUNK;
    // ---- stage chunk B/C tiles (8KB each, fully contiguous) into LDS ----
#if USE_TDM
    if (wave == 0) {
      tdm_load_1d_f32((u32)(uintptr_t)&sB[0], Bc + base * DSTATE, CHUNK * DSTATE);
      tdm_load_1d_f32((u32)(uintptr_t)&sC[0], Cc + base * DSTATE, CHUNK * DSTATE);
    }
#else
    {
      const float4* Bg = (const float4*)(Bc + base * DSTATE);
      const float4* Cg = (const float4*)(Cc + base * DSTATE);
      for (int i = t; i < (CHUNK * DSTATE) / 4; i += 256) {
        ((float4*)sB)[i] = Bg[i];
        ((float4*)sC)[i] = Cg[i];
      }
    }
#endif
    for (int i = t; i < CHUNK * HD; i += 256) {
      int l = i >> 3, p = i & 7;
      sx[i] = xh[(base + l) * DIN + h * HD + p] * dtb[(base + l) * NH + h];
    }
    if (t == 0) {
      float run = 0.f;
      for (int l = 0; l < CHUNK; ++l) { run += dA[(base + l) * NH + h]; sac[l] = run; }
    }
    // prefetch next chunk (global_prefetch_b8)
    if (c + 1 < NCHUNK && t < CHUNK) {
      const long nb = base + CHUNK;
      __builtin_prefetch(&Bc[(nb + t) * DSTATE], 0, 1);
      __builtin_prefetch(&Cc[(nb + t) * DSTATE], 0, 1);
    }
#if USE_TDM
    if (wave == 0) __builtin_amdgcn_s_wait_tensorcnt(0);
#endif
    __syncthreads();

    // G[l][s] = sum_n C[l][n]*B[s][n] : 2x2 tiles, K=64 -> waves 0..3
    if (wave < 4) {
      const int tm = wave >> 1, tn = wave & 1;
      v8f acc = {};
      #pragma unroll
      for (int ks = 0; ks < 2; ++ks) {
        v16bf a, b;
        #pragma unroll
        for (int i = 0; i < 16; ++i) {
          a[i] = (__bf16)sC[(tm * 16 + a_row(lane)) * DSTATE + a_col(lane, i) + ks * 32];
          b[i] = (__bf16)sB[(tn * 16 + b_col(lane)) * DSTATE + b_row(lane, i) + ks * 32];
        }
        acc = wmma_bf16(a, b, acc);
      }
      #pragma unroll
      for (int r = 0; r < 8; ++r)
        sG[(tm * 16 + d_row(lane, r)) * CHUNK + tn * 16 + d_col(lane)] = acc[r];
    }
    __syncthreads();

    // Y[l][p] = sum_{s<=l} G*exp(a_l-a_s)*x[s][p] + exp(a_l)*C[l].S[p]
    {
      const int l = t >> 3, p = t & 7;
      const float al = sac[l];
      float y = 0.f;
      for (int s = 0; s <= l; ++s)
        y += sG[l * CHUNK + s] * __expf(al - sac[s]) * sx[s * HD + p];
      float dot = 0.f;
      for (int n = 0; n < DSTATE; ++n)
        dot += sC[l * DSTATE + n] * sS[p * DSTATE + n];
      y += __expf(al) * dot;
      ys[(base + l) * DIN + h * HD + p] = y;
    }
    __syncthreads();

    // state update: S = exp(a_last)*S + sum_l exp(a_last-a_l) x[l][p] B[l][n]
    const float alast = sac[CHUNK - 1];
    for (int i = t; i < HD * DSTATE; i += 256) {
      int p = i >> 6, n = i & 63;
      float s = __expf(alast) * sS[i];
      for (int l = 0; l < CHUNK; ++l)
        s += __expf(alast - sac[l]) * sx[l * HD + p] * sB[l * DSTATE + n];
      sS[i] = s;
    }
    __syncthreads();
  }
}

// ---------------- K5: + x*D, gate silu(z), RMSNorm -> bf16 A-frag order -----
__global__ void k5_gate(const float* __restrict__ ys, const float* __restrict__ xh,
                        const float* __restrict__ zx, const float* __restrict__ Dp,
                        const float* __restrict__ rms_w, __bf16* __restrict__ ybf) {
  const int lane = threadIdx.x & 31, wave = threadIdx.x >> 5;
  const long g = (long)blockIdx.x * 8 + wave;   // token flat over NB*HW
  const long T = g >> 4; const int m = (int)(g & 15);
  float yv[2]; float ss = 0.f;
  #pragma unroll
  for (int j = 0; j < 2; ++j) {
    int d = lane * 2 + j;
    float y = ys[g * DIN + d] + xh[g * DIN + d] * Dp[d >> 3];
    float z = zx[g * DPROJ + d];
    y *= silu_f(z);
    yv[j] = y; ss += y * y;
  }
  ss = warp_allreduce_sum(ss);
  const float r = rsqrtf(ss * (1.f / DIN) + 1e-5f);
  #pragma unroll
  for (int j = 0; j < 2; ++j) {
    int d = lane * 2 + j;
    int ks = d >> 5;
    ybf[T * 1024 + ks * 512 + afrag_pos(m, d & 31)] = (__bf16)(yv[j] * r * rms_w[d]);
  }
}

// ---------------- K6: out-proj (64->32) WMMA + skip + recombine -------------
__global__ void k6_outproj(const __bf16* __restrict__ ybf, const __bf16* __restrict__ Woutb,
                           const float* __restrict__ xs_f, const float* __restrict__ skip,
                           float* __restrict__ ymc) {
  const int lane = threadIdx.x & 31, wave = threadIdx.x >> 5;
  const long t0 = (long)blockIdx.x * 128 + wave * 16;
  const long T0 = t0 >> 4;
  const float sk = skip[0];
  const v16bf a0 = *(const v16bf*)(ybf + T0 * 1024 + lane * 16);
  const v16bf a1 = *(const v16bf*)(ybf + T0 * 1024 + 512 + lane * 16);
  #pragma unroll
  for (int nt = 0; nt < 2; ++nt) {
    const v16bf b0 = *(const v16bf*)(Woutb + (0 * 2 + nt) * 512 + lane * 16);
    const v16bf b1 = *(const v16bf*)(Woutb + (1 * 2 + nt) * 512 + lane * 16);
    v8f acc = {};
    acc = wmma_bf16(a0, b0, acc);
    acc = wmma_bf16(a1, b1, acc);
    #pragma unroll
    for (int r = 0; r < 8; ++r) {
      const long g = t0 + d_row(lane, r);
      const int d = nt * 16 + d_col(lane);
      const float val = acc[r] + sk * xs_f[g * DM + d];
      const long bi = g / HW; const int l = (int)(g % HW);
      const int s = (int)(bi >> 3), bb = (int)(bi & 7);   // bi = s*B0 + bb
      ymc[((long)bb * HW + l) * CIN + s * DM + d] = val;
    }
  }
}

// ---------------- K7: LayerNorm over 256 -> bf16 A-frag order ---------------
__global__ void k7_ln2(const float* __restrict__ ymc, const float* __restrict__ ln_w,
                       const float* __restrict__ ln_b, __bf16* __restrict__ ynbf) {
  const int lane = threadIdx.x & 31, wave = threadIdx.x >> 5;
  const long g = (long)blockIdx.x * 8 + wave;   // token flat over B0*HW
  const long T = g >> 4; const int m = (int)(g & 15);
  float v[8]; float sum = 0.f;
  #pragma unroll
  for (int j = 0; j < 8; ++j) { v[j] = ymc[g * CIN + lane * 8 + j]; sum += v[j]; }
  sum = warp_allreduce_sum(sum);
  const float mu = sum * (1.f / 256.f);
  float sq = 0.f;
  #pragma unroll
  for (int j = 0; j < 8; ++j) { float d = v[j] - mu; sq += d * d; }
  sq = warp_allreduce_sum(sq);
  const float inv = rsqrtf(sq * (1.f / 256.f) + 1e-5f);
  #pragma unroll
  for (int j = 0; j < 8; ++j) {
    int cc = lane * 8 + j;
    float o = (v[j] - mu) * inv * ln_w[cc] + ln_b[cc];
    int ks = cc >> 5;
    ynbf[T * 4096 + (long)ks * 512 + afrag_pos(m, cc & 31)] = (__bf16)o;
  }
}

// ---------------- K8: final GEMM 256x256 via WMMA, transposed store ---------
__global__ void k8_proj(const __bf16* __restrict__ ynbf, const __bf16* __restrict__ Pwb,
                        const float* __restrict__ proj_b, float* __restrict__ out) {
  const int lane = threadIdx.x & 31, wave = threadIdx.x >> 5;
  const long g0 = (long)blockIdx.x * 128 + wave * 16;   // token flat over B0*HW
  const long T0 = g0 >> 4;
  v16bf aa[8];
  #pragma unroll
  for (int ks = 0; ks < 8; ++ks)
    aa[ks] = *(const v16bf*)(ynbf + T0 * 4096 + ks * 512 + lane * 16);
  for (int nt = 0; nt < 16; ++nt) {
    v8f acc = {};
    #pragma unroll
    for (int ks = 0; ks < 8; ++ks) {
      const v16bf b = *(const v16bf*)(Pwb + (nt * 8 + ks) * 512 + lane * 16);
      acc = wmma_bf16(aa[ks], b, acc);
    }
    const int n = nt * 16 + d_col(lane);
    const float pb = proj_b[n];
    #pragma unroll
    for (int r = 0; r < 8; ++r) {
      const long g = g0 + d_row(lane, r);
      const long bb = g / HW; const int l = (int)(g % HW);
      out[(bb * DOUT + n) * (long)HW + l] = acc[r] + pb;
    }
  }
}

// ---------------- launch ----------------------------------------------------
extern "C" void kernel_launch(void* const* d_in, const int* in_sizes, int n_in,
                              void* d_out, int out_size, void* d_ws, size_t ws_size,
                              hipStream_t stream) {
  (void)in_sizes; (void)n_in; (void)out_size; (void)ws_size;
  const float* x       = (const float*)d_in[0];
  const float* ln_w    = (const float*)d_in[1];
  const float* ln_b    = (const float*)d_in[2];
  const float* skip    = (const float*)d_in[3];
  const float* proj_w  = (const float*)d_in[4];
  const float* proj_b  = (const float*)d_in[5];
  const float* W_in    = (const float*)d_in[6];
  const float* conv_w  = (const float*)d_in[7];
  const float* conv_b  = (const float*)d_in[8];
  const float* dt_bias = (const float*)d_in[9];
  const float* A_log   = (const float*)d_in[10];
  const float* Dp      = (const float*)d_in[11];
  const float* rms_w   = (const float*)d_in[12];
  const float* W_out   = (const float*)d_in[13];
  float* out = (float*)d_out;

  char* ws = (char*)d_ws;
  float*  xs_f  = (float*) (ws + OFF_XSF);
  __bf16* xs_bf = (__bf16*)(ws + OFF_XSB);
  float*  zx    = (float*) (ws + OFF_ZX);
  float*  xh    = (float*) (ws + OFF_XH);
  float*  Bc    = (float*) (ws + OFF_BC);
  float*  Cc    = (float*) (ws + OFF_CC);
  float*  dtb   = (float*) (ws + OFF_DT);
  float*  dA    = (float*) (ws + OFF_DA);
  float*  ys    = (float*) (ws + OFF_YS);
  __bf16* ybf   = (__bf16*)(ws + OFF_YBF);
  float*  ymc   = (float*) (ws + OFF_YMC);
  __bf16* ynbf  = (__bf16*)(ws + OFF_YNB);
  __bf16* Winb  = (__bf16*)(ws + OFF_WIN);
  __bf16* Woutb = (__bf16*)(ws + OFF_WOU);
  __bf16* Pwb   = (__bf16*)(ws + OFF_PW);

  k0_prep   <<<dim3(256), 256, 0, stream>>>(W_in, W_out, proj_w, Winb, Woutb, Pwb);
  k1_ln1    <<<dim3(HW / 8, B0), 256, 0, stream>>>(x, ln_w, ln_b, xs_f, xs_bf);
  k2_inproj <<<dim3((NB * HW) / 128), 256, 0, stream>>>(xs_bf, Winb, zx);
  k3_conv   <<<dim3(HW / 64, NB), 256, 0, stream>>>(zx, conv_w, conv_b, dt_bias, A_log,
                                                    xh, Bc, Cc, dtb, dA);
  k4_ssd    <<<dim3(NB, NH), 256, 0, stream>>>(Bc, Cc, xh, dtb, dA, ys);
  k5_gate   <<<dim3((NB * HW) / 8), 256, 0, stream>>>(ys, xh, zx, Dp, rms_w, ybf);
  k6_outproj<<<dim3((NB * HW) / 128), 256, 0, stream>>>(ybf, Woutb, xs_f, skip, ymc);
  k7_ln2    <<<dim3((B0 * HW) / 8), 256, 0, stream>>>(ymc, ln_w, ln_b, ynbf);
  k8_proj   <<<dim3((B0 * HW) / 128), 256, 0, stream>>>(ynbf, Pwb, proj_b, out);
}
